// NystromEncoder_48567490183282
// MI455X (gfx1250) — matrix operand order repaired and checked by reading, not compiled, not run on hardware
//
#include <hip/hip_runtime.h>
#include <math.h>

// ---------------------------------------------------------------------------
// Types for CDNA5 WMMA (wave32)
// ---------------------------------------------------------------------------
typedef __attribute__((ext_vector_type(16))) __bf16 v16bf;
typedef __attribute__((ext_vector_type(8)))  float  v8f;
typedef __attribute__((ext_vector_type(8)))  __bf16 bf8;

#define N_TOK 16384
#define DMODEL 512
#define NHEAD 8
#define DHEAD 64
#define MLM   256
#define LGRP  64
#define FFDIM 2048
#define KCONV 33
#define N_P   2048
#define N_H   14336

__device__ __forceinline__ __bf16 f2bf(float f) {
    unsigned u = __float_as_uint(f);
    u += 0x7FFFu + ((u >> 16) & 1u);   // round-to-nearest-even
    unsigned short s = (unsigned short)(u >> 16);
    return __builtin_bit_cast(__bf16, s);
}
__device__ __forceinline__ float bf2f(__bf16 h) {
    unsigned short s = __builtin_bit_cast(unsigned short, h);
    return __uint_as_float(((unsigned)s) << 16);
}

// ---------------------------------------------------------------------------
// Elementwise helpers
// ---------------------------------------------------------------------------
__global__ void k_f32_to_bf16(const float* __restrict__ in, __bf16* __restrict__ out,
                              long long n) {
    long long i = (long long)blockIdx.x * blockDim.x + threadIdx.x;
    if (i < n) out[i] = f2bf(in[i]);
}

// Transpose + convert: src[R][C] f32 -> dst[C][R] bf16, batched via blockIdx.z
__global__ void k_f32_to_bf16_T(const float* __restrict__ src, __bf16* __restrict__ dst,
                                int R, int C) {
    __shared__ float t[32][33];
    src += (size_t)blockIdx.z * R * C;
    dst += (size_t)blockIdx.z * R * C;
    int r0 = blockIdx.y * 32, c0 = blockIdx.x * 32;
    int tx = threadIdx.x & 31, ty = threadIdx.x >> 5;    // 32 x 8
    for (int j = 0; j < 32; j += 8)
        t[ty + j][tx] = src[(size_t)(r0 + ty + j) * C + c0 + tx];
    __syncthreads();
    for (int j = 0; j < 32; j += 8)
        dst[(size_t)(c0 + ty + j) * R + r0 + tx] = f2bf(t[tx][ty + j]);
}

__global__ void k_zero_f32(float* __restrict__ p, long long n) {
    long long i = (long long)blockIdx.x * blockDim.x + threadIdx.x;
    if (i < n) p[i] = 0.0f;
}

// y = d*I + s*x  over H batches of 256x256
__global__ void k_diag_axpy(const float* __restrict__ x, float* __restrict__ y,
                            float d, float s) {
    long long i = (long long)blockIdx.x * blockDim.x + threadIdx.x;
    int e = (int)(i & (MLM * MLM - 1));
    int r = e >> 8, c = e & 255;
    y[i] = ((r == c) ? d : 0.0f) + s * x[i];
}

// ---------------------------------------------------------------------------
// LayerNorm: one block (256 threads) per token row of 512, bf16 output
// ---------------------------------------------------------------------------
__global__ void k_layernorm(const float* __restrict__ x, const float* __restrict__ w,
                            const float* __restrict__ b, __bf16* __restrict__ out) {
    size_t row = blockIdx.x;
    x += row * DMODEL; out += row * DMODEL;
    int t = threadIdx.x;
    float a0 = x[t], a1 = x[t + 256];
    __shared__ float red[256];
    red[t] = a0 + a1; __syncthreads();
    for (int s = 128; s; s >>= 1) { if (t < s) red[t] += red[t + s]; __syncthreads(); }
    float mu = red[0] * (1.0f / DMODEL); __syncthreads();
    float d0 = a0 - mu, d1 = a1 - mu;
    red[t] = d0 * d0 + d1 * d1; __syncthreads();
    for (int s = 128; s; s >>= 1) { if (t < s) red[t] += red[t + s]; __syncthreads(); }
    float rstd = rsqrtf(red[0] * (1.0f / DMODEL) + 1e-5f);
    out[t]       = f2bf(d0 * rstd * w[t]       + b[t]);
    out[t + 256] = f2bf(d1 * rstd * w[t + 256] + b[t + 256]);
}

// ---------------------------------------------------------------------------
// Row softmax: one block per row; streaming 3-pass; writes bf16 and/or f32
// ---------------------------------------------------------------------------
__global__ void k_softmax(const float* __restrict__ S, __bf16* __restrict__ ob,
                          float* __restrict__ of, int cols) {
    size_t row = blockIdx.x;
    S += row * (size_t)cols;
    if (ob) ob += row * (size_t)cols;
    if (of) of += row * (size_t)cols;
    int t = threadIdx.x;
    __shared__ float red[256];
    float m = -3.4e38f;
    for (int c = t; c < cols; c += 256) m = fmaxf(m, S[c]);
    red[t] = m; __syncthreads();
    for (int s = 128; s; s >>= 1) { if (t < s) red[t] = fmaxf(red[t], red[t + s]); __syncthreads(); }
    float mx = red[0]; __syncthreads();
    float sum = 0.0f;
    for (int c = t; c < cols; c += 256) sum += expf(S[c] - mx);
    red[t] = sum; __syncthreads();
    for (int s = 128; s; s >>= 1) { if (t < s) red[t] += red[t + s]; __syncthreads(); }
    float inv = 1.0f / red[0];
    for (int c = t; c < cols; c += 256) {
        float e = expf(S[c] - mx) * inv;
        if (ob) ob[c] = f2bf(e);
        if (of) of[c] = e;
    }
}

// ---------------------------------------------------------------------------
// Landmark mean pooling: q_l[h][m][c] and k_l[h][m][c] (k_l is B^T for s1/s2)
// ---------------------------------------------------------------------------
__global__ void k_landmarks(const __bf16* __restrict__ q, const __bf16* __restrict__ k,
                            __bf16* __restrict__ ql, __bf16* __restrict__ kl) {
    int idx = blockIdx.x * blockDim.x + threadIdx.x;   // H*MLM*DHEAD
    int c = idx % DHEAD;
    int m = (idx / DHEAD) % MLM;
    int h = idx / (DHEAD * MLM);
    int d = h * DHEAD + c;
    float sq = 0.0f, sk = 0.0f;
    for (int i = 0; i < LGRP; ++i) {
        size_t n = (size_t)m * LGRP + i;
        sq += bf2f(q[n * DMODEL + d]);
        sk += bf2f(k[n * DMODEL + d]);
    }
    size_t o = ((size_t)h * MLM + m) * DHEAD + c;
    ql[o] = f2bf(sq * (1.0f / LGRP));
    kl[o] = f2bf(sk * (1.0f / LGRP));
}

// ---------------------------------------------------------------------------
// Tiled bf16 transpose: src[N_TOK][DMODEL] -> dst[DMODEL][N_TOK]
// ---------------------------------------------------------------------------
__global__ void k_transpose(const __bf16* __restrict__ src, __bf16* __restrict__ dst) {
    __shared__ __bf16 t[32][33];
    int n0 = blockIdx.x * 32, d0 = blockIdx.y * 32;
    int tx = threadIdx.x & 31, ty = threadIdx.x >> 5;    // 32 x 8
    for (int j = 0; j < 32; j += 8)
        t[ty + j][tx] = src[(size_t)(n0 + ty + j) * DMODEL + d0 + tx];
    __syncthreads();
    for (int j = 0; j < 32; j += 8)
        dst[(size_t)(d0 + ty + j) * N_TOK + n0 + tx] = t[tx][ty + j];
}

// ---------------------------------------------------------------------------
// Depthwise residual conv (K=33) along n, shared over dh: attn += conv(v)
// ---------------------------------------------------------------------------
__global__ void k_resconv(const __bf16* __restrict__ v, const float* __restrict__ w,
                          float* __restrict__ attn) {
    long long idx = (long long)blockIdx.x * blockDim.x + threadIdx.x;
    int n = (int)(idx >> 9);
    int d = (int)(idx & 511);
    int h = d >> 6;
    float acc = 0.0f;
    for (int t = 0; t < KCONV; ++t) {
        int nn = n + t - (KCONV / 2);
        if (nn >= 0 && nn < N_TOK)
            acc += w[h * KCONV + t] * bf2f(v[(size_t)nn * DMODEL + d]);
    }
    attn[idx] += acc;
}

// ---------------------------------------------------------------------------
// Newton-Schulz pinv helpers (fp32, per-head 256x256)
// ---------------------------------------------------------------------------
__global__ void k_pinv_init(const float* __restrict__ a2, float* __restrict__ z) {
    int h = blockIdx.x;
    a2 += (size_t)h * MLM * MLM; z += (size_t)h * MLM * MLM;
    int t = threadIdx.x;
    float rs = 0.0f, cs = 0.0f;
    for (int j = 0; j < MLM; ++j) {
        rs += fabsf(a2[(size_t)t * MLM + j]);
        cs += fabsf(a2[(size_t)j * MLM + t]);
    }
    __shared__ float red[256];
    red[t] = rs; __syncthreads();
    for (int s = 128; s; s >>= 1) { if (t < s) red[t] = fmaxf(red[t], red[t + s]); __syncthreads(); }
    float mr = red[0]; __syncthreads();
    red[t] = cs; __syncthreads();
    for (int s = 128; s; s >>= 1) { if (t < s) red[t] = fmaxf(red[t], red[t + s]); __syncthreads(); }
    float sc = 1.0f / (mr * red[0]);
    for (int j = 0; j < MLM; ++j)
        z[(size_t)t * MLM + j] = a2[(size_t)j * MLM + t] * sc;   // transpose * scale
}

// C = diag*I + scl*(A@B), batched over heads (blockIdx.z), 256x256 fp32
__global__ void k_mm256(const float* __restrict__ A, const float* __restrict__ B,
                        float* __restrict__ C, float diag, float scl) {
    int h = blockIdx.z;
    A += (size_t)h * MLM * MLM; B += (size_t)h * MLM * MLM; C += (size_t)h * MLM * MLM;
    __shared__ float As[64][17];
    __shared__ float Bs[16][65];
    int tid = threadIdx.x;
    int tx = tid & 15, ty = tid >> 4;
    int r0 = blockIdx.y * 64, c0 = blockIdx.x * 64;
    float acc[4][4] = {};
    for (int k0 = 0; k0 < MLM; k0 += 16) {
        for (int i = tid; i < 64 * 16; i += 256) {
            int r = i >> 4, c = i & 15;
            As[r][c] = A[(size_t)(r0 + r) * MLM + k0 + c];
        }
        for (int i = tid; i < 16 * 64; i += 256) {
            int r = i >> 6, c = i & 63;
            Bs[r][c] = B[(size_t)(k0 + r) * MLM + c0 + c];
        }
        __syncthreads();
#pragma unroll
        for (int kk = 0; kk < 16; ++kk) {
            float av[4], bv[4];
#pragma unroll
            for (int i = 0; i < 4; ++i) { av[i] = As[ty * 4 + i][kk]; bv[i] = Bs[kk][tx * 4 + i]; }
#pragma unroll
            for (int i = 0; i < 4; ++i)
#pragma unroll
                for (int j = 0; j < 4; ++j) acc[i][j] += av[i] * bv[j];
        }
        __syncthreads();
    }
#pragma unroll
    for (int i = 0; i < 4; ++i)
#pragma unroll
        for (int j = 0; j < 4; ++j) {
            int r = r0 + ty * 4 + i, c = c0 + tx * 4 + j;
            C[(size_t)r * MLM + c] = ((r == c) ? diag : 0.0f) + scl * acc[i][j];
        }
}

// ---------------------------------------------------------------------------
// WMMA fragment loaders from LDS (both tiles staged row-major-in-K, stride 40)
//   A: 16-bit A 16x32 ISA layout: lanes 0-15 K {0..7,16..23}, lanes 16-31 +8
//   B: lane n holds its K-column, staged as B^T rows: lanes 0-15 K {0..7,16..23},
//      lanes 16-31 +8  -> two b128 LDS loads each
// ---------------------------------------------------------------------------
__device__ __forceinline__ v16bf load_fragA(const __bf16* base, int rbase, int lane) {
    int half = lane >> 4;
    const __bf16* p = base + (size_t)(rbase + (lane & 15)) * 40 + half * 8;
    bf8 lo = *reinterpret_cast<const bf8*>(p);
    bf8 hi = *reinterpret_cast<const bf8*>(p + 16);
    v16bf f;
#pragma unroll
    for (int j = 0; j < 8; ++j) { f[j] = lo[j]; f[j + 8] = hi[j]; }
    return f;
}
__device__ __forceinline__ v16bf load_fragB(const __bf16* base, int nbase, int lane) {
    int half = lane >> 4;
    const __bf16* p = base + (size_t)(nbase + (lane & 15)) * 40 + half * 8;
    bf8 lo = *reinterpret_cast<const bf8*>(p);
    bf8 hi = *reinterpret_cast<const bf8*>(p + 16);
    v16bf f;
#pragma unroll
    for (int j = 0; j < 8; ++j) { f[j] = lo[j]; f[j + 8] = hi[j]; }
    return f;
}

#define WMMA_BF16(a, b, c) \
    __builtin_amdgcn_wmma_f32_16x16x32_bf16(false, (a), false, (b), (short)0, (c), false, false)

// ---------------------------------------------------------------------------
// Generic batched bf16 WMMA GEMM:  C[M,N] (+epilogue) = A[M,K] @ B[K,N]
//   B is supplied TRANSPOSED (BT[N][K] row-major) so both LDS stages and both
//   fragment loads are fully vectorized (b128).
//   - wave tile 32x64 -> 2x4 WMMA per 32-K step
//   - blockIdx.z = batch * ksplit + kseg; split-K path uses atomicAdd into Cf
//   - epilogue: val = acc*scale + bias[n] + resid[m,n]; optional exact GELU;
//     writes fp32 (Cf) and/or bf16 (Cb); cbT writes Cb transposed [N][M]
// ---------------------------------------------------------------------------
template <int BM, int BN>
__global__ __launch_bounds__((BM / 32) * (BN / 64) * 32)
void k_gemm(const __bf16* __restrict__ A, int lda, long long sA,
            const __bf16* __restrict__ BT, int ldb, long long sB,
            float* __restrict__ Cf, __bf16* __restrict__ Cb, int ldc, long long sC,
            const float* __restrict__ bias, const float* __restrict__ resid, int ldr,
            int M, int N, int K, int ksplit, float scale, int do_gelu, int cbT) {
    constexpr int WAVESN = BN / 64;
    constexpr int NWAVE = (BM / 32) * (BN / 64);
    constexpr int NT = NWAVE * 32;
    __shared__ __bf16 As[BM * 40];
    __shared__ __bf16 Bs[BN * 40];

    const int tid = threadIdx.x;
    const int wave = tid >> 5;
    const int lane = tid & 31;
    const int batch = blockIdx.z / ksplit;
    const int kseg = blockIdx.z % ksplit;
    A += (long long)batch * sA;
    BT += (long long)batch * sB;
    if (Cf) Cf += (long long)batch * sC;
    if (Cb) Cb += (long long)batch * sC;

    const int m0 = blockIdx.y * BM;
    const int n0 = blockIdx.x * BN;
    const int wm = (wave / WAVESN) * 32;
    const int wn = (wave % WAVESN) * 64;

    v8f acc[2][4] = {};

    const int kper = K / ksplit;
    const int kbeg = kseg * kper;
    const int kend = kbeg + kper;

    for (int k0 = kbeg; k0 < kend; k0 += 32) {
        // stage A tile (BM x 32) and BT tile (BN x 32), all vector b128
        for (int idx = tid * 8; idx < BM * 32; idx += NT * 8) {
            int r = idx >> 5, c = idx & 31;
            bf8 v = *reinterpret_cast<const bf8*>(A + (long long)(m0 + r) * lda + k0 + c);
            *reinterpret_cast<bf8*>(&As[r * 40 + c]) = v;
        }
        for (int idx = tid * 8; idx < BN * 32; idx += NT * 8) {
            int r = idx >> 5, c = idx & 31;
            bf8 v = *reinterpret_cast<const bf8*>(BT + (long long)(n0 + r) * ldb + k0 + c);
            *reinterpret_cast<bf8*>(&Bs[r * 40 + c]) = v;
        }
        if (k0 + 32 < kend) {
            __builtin_prefetch(A + (long long)(m0 + (tid & (BM - 1))) * lda + k0 + 32, 0, 1);
            __builtin_prefetch(BT + (long long)(n0 + (tid & (BN - 1))) * ldb + k0 + 32, 0, 1);
        }
        __syncthreads();

        v16bf af[2];
        af[0] = load_fragA(As, wm + 0, lane);
        af[1] = load_fragA(As, wm + 16, lane);
        v16bf bfr[4];
#pragma unroll
        for (int nt = 0; nt < 4; ++nt) bfr[nt] = load_fragB(Bs, wn + nt * 16, lane);
#pragma unroll
        for (int mt = 0; mt < 2; ++mt)
#pragma unroll
            for (int nt = 0; nt < 4; ++nt)
                acc[mt][nt] = WMMA_BF16(af[mt], bfr[nt], acc[mt][nt]);
        __syncthreads();
    }

    // epilogue (C layout: lanes 0-15 -> n=lane, M=r; lanes 16-31 -> n=lane-16, M=r+8)
    const int l15 = lane & 15;
    const int mo = (lane >> 4) * 8;
#pragma unroll
    for (int mt = 0; mt < 2; ++mt)
#pragma unroll
        for (int nt = 0; nt < 4; ++nt) {
            int n = n0 + wn + nt * 16 + l15;
            if (n >= N) continue;
#pragma unroll
            for (int r = 0; r < 8; ++r) {
                int m = m0 + wm + mt * 16 + mo + r;
                if (m >= M) continue;
                long long idx = (long long)m * ldc + n;
                float val = acc[mt][nt][r] * scale;
                if (ksplit > 1) { atomicAdd(Cf + idx, val); continue; }
                if (bias) val += bias[n];
                if (resid) val += resid[(long long)m * ldr + n];
                if (do_gelu) val = 0.5f * val * (1.0f + erff(val * 0.70710678118654752f));
                if (Cf) Cf[idx] = val;
                if (Cb) Cb[cbT ? ((long long)n * ldc + m) : idx] = f2bf(val);
            }
        }
}

// ---------------------------------------------------------------------------
// Host orchestration
// ---------------------------------------------------------------------------
extern "C" void kernel_launch(void* const* d_in, const int* in_sizes, int n_in,
                              void* d_out, int out_size, void* d_ws, size_t ws_size,
                              hipStream_t stream) {
    (void)in_sizes; (void)n_in; (void)out_size; (void)ws_size;

    const float* p_tok = (const float*)d_in[0];
    const float* h_tok = (const float*)d_in[1];

    // workspace arena
    size_t off = 0;
    char* base = (char*)d_ws;
    auto alloc = [&](size_t bytes) -> char* {
        char* p = base + off;
        off += (bytes + 255) & ~(size_t)255;
        return p;
    };
    float*  x      = (float*) alloc((size_t)N_TOK * DMODEL * 4);
    __bf16* xln    = (__bf16*)alloc((size_t)N_TOK * DMODEL * 2);
    __bf16* qbf    = (__bf16*)alloc((size_t)N_TOK * DMODEL * 2);
    __bf16* kbf    = (__bf16*)alloc((size_t)N_TOK * DMODEL * 2);
    __bf16* vbf    = (__bf16*)alloc((size_t)N_TOK * DMODEL * 2);
    __bf16* vT     = (__bf16*)alloc((size_t)DMODEL * N_TOK * 2);
    __bf16* qlbf   = (__bf16*)alloc((size_t)NHEAD * MLM * DHEAD * 2);
    __bf16* klbf   = (__bf16*)alloc((size_t)NHEAD * MLM * DHEAD * 2);
    float*  scores = (float*) alloc((size_t)NHEAD * N_TOK * MLM * 4);  // s1, then s3, then aliased ff
    __bf16* a1bf   = (__bf16*)alloc((size_t)NHEAD * N_TOK * MLM * 2);
    __bf16* a3bf   = (__bf16*)alloc((size_t)NHEAD * MLM * N_TOK * 2);
    float*  s2     = (float*) alloc((size_t)NHEAD * MLM * MLM * 4);
    float*  a2     = (float*) alloc((size_t)NHEAD * MLM * MLM * 4);
    float*  zA     = (float*) alloc((size_t)NHEAD * MLM * MLM * 4);
    float*  zB     = (float*) alloc((size_t)NHEAD * MLM * MLM * 4);
    float*  xz     = (float*) alloc((size_t)NHEAD * MLM * MLM * 4);
    float*  y1     = (float*) alloc((size_t)NHEAD * MLM * MLM * 4);
    float*  y2     = (float*) alloc((size_t)NHEAD * MLM * MLM * 4);
    float*  y3     = (float*) alloc((size_t)NHEAD * MLM * MLM * 4);
    float*  a3v    = (float*) alloc((size_t)NHEAD * MLM * DHEAD * 4);
    __bf16* zbf    = (__bf16*)alloc((size_t)NHEAD * MLM * MLM * 2);
    __bf16* a3vT   = (__bf16*)alloc((size_t)NHEAD * DHEAD * MLM * 2);
    __bf16* w2T    = (__bf16*)alloc((size_t)NHEAD * DHEAD * MLM * 2);
    float*  attn   = (float*) alloc((size_t)N_TOK * DMODEL * 4);
    __bf16* attnbf = (__bf16*)alloc((size_t)N_TOK * DMODEL * 2);
    __bf16* wqkvT  = (__bf16*)alloc((size_t)DMODEL * 3 * DMODEL * 2);
    __bf16* woutT  = (__bf16*)alloc((size_t)DMODEL * DMODEL * 2);
    __bf16* wff1T  = (__bf16*)alloc((size_t)DMODEL * FFDIM * 2);
    __bf16* wff2T  = (__bf16*)alloc((size_t)FFDIM * DMODEL * 2);
    __bf16* ffbf   = (__bf16*)scores;   // alias: scores dead by FF time

    auto cvt = [&](const float* src, __bf16* dst, long long n) {
        k_f32_to_bf16<<<dim3((unsigned)((n + 255) / 256)), 256, 0, stream>>>(src, dst, n);
    };
    auto cvtT = [&](const float* src, __bf16* dst, int R, int C, int batch) {
        k_f32_to_bf16_T<<<dim3(C / 32, R / 32, batch), 256, 0, stream>>>(src, dst, R, C);
    };
    auto gemm128 = [&](const __bf16* A, int lda, long long sA,
                       const __bf16* BT, int ldb, long long sB,
                       float* Cf, __bf16* Cb, int ldc, long long sC,
                       const float* bias, const float* resid, int ldr,
                       int M, int N, int K, int batch, int ksplit, float scale, int gelu, int cbT) {
        dim3 g(N / 128, M / 64, batch * ksplit);
        k_gemm<64, 128><<<g, 128, 0, stream>>>(A, lda, sA, BT, ldb, sB, Cf, Cb, ldc, sC,
                                               bias, resid, ldr, M, N, K, ksplit, scale, gelu, cbT);
    };
    auto gemm64 = [&](const __bf16* A, int lda, long long sA,
                      const __bf16* BT, int ldb, long long sB,
                      float* Cf, __bf16* Cb, int ldc, long long sC,
                      const float* bias, const float* resid, int ldr,
                      int M, int N, int K, int batch, int ksplit, float scale, int gelu, int cbT) {
        dim3 g(N / 64, M / 64, batch * ksplit);
        k_gemm<64, 64><<<g, 64, 0, stream>>>(A, lda, sA, BT, ldb, sB, Cf, Cb, ldc, sC,
                                             bias, resid, ldr, M, N, K, ksplit, scale, gelu, cbT);
    };

    // x = concat(p_tokens, h_tokens)
    hipMemcpyAsync(x, p_tok, (size_t)N_P * DMODEL * 4, hipMemcpyDeviceToDevice, stream);
    hipMemcpyAsync(x + (size_t)N_P * DMODEL, h_tok, (size_t)N_H * DMODEL * 4,
                   hipMemcpyDeviceToDevice, stream);

    for (int L = 0; L < 2; ++L) {
        const float* ln1w = (const float*)d_in[2]  + (size_t)L * DMODEL;
        const float* ln1b = (const float*)d_in[3]  + (size_t)L * DMODEL;
        const float* wqkv = (const float*)d_in[4]  + (size_t)L * DMODEL * 3 * DMODEL;
        const float* wout = (const float*)d_in[5]  + (size_t)L * DMODEL * DMODEL;
        const float* bout = (const float*)d_in[6]  + (size_t)L * DMODEL;
        const float* cw   = (const float*)d_in[7]  + (size_t)L * NHEAD * KCONV;
        const float* ln2w = (const float*)d_in[8]  + (size_t)L * DMODEL;
        const float* ln2b = (const float*)d_in[9]  + (size_t)L * DMODEL;
        const float* wff1 = (const float*)d_in[10] + (size_t)L * DMODEL * FFDIM;
        const float* bff1 = (const float*)d_in[11] + (size_t)L * FFDIM;
        const float* wff2 = (const float*)d_in[12] + (size_t)L * FFDIM * DMODEL;
        const float* bff2 = (const float*)d_in[13] + (size_t)L * DMODEL;

        // weights -> bf16, transposed to [N][K] for the GEMM B-operand
        cvtT(wqkv, wqkvT, DMODEL, 3 * DMODEL, 1);   // [1536][512]
        cvtT(wout, woutT, DMODEL, DMODEL, 1);       // [512][512]
        cvtT(wff1, wff1T, DMODEL, FFDIM, 1);        // [2048][512]
        cvtT(wff2, wff2T, FFDIM, DMODEL, 1);        // [512][2048]

        // LN1 -> bf16
        k_layernorm<<<N_TOK, 256, 0, stream>>>(x, ln1w, ln1b, xln);

        // q (scaled dh^-0.5), k, v  : three 16384x512x512 WMMA GEMMs
        gemm128(xln, DMODEL, 0, wqkvT + 0 * DMODEL * DMODEL, DMODEL, 0, nullptr, qbf, DMODEL, 0,
                nullptr, nullptr, 0, N_TOK, DMODEL, DMODEL, 1, 1, 0.125f, 0, 0);
        gemm128(xln, DMODEL, 0, wqkvT + 1 * DMODEL * DMODEL, DMODEL, 0, nullptr, kbf, DMODEL, 0,
                nullptr, nullptr, 0, N_TOK, DMODEL, DMODEL, 1, 1, 1.0f, 0, 0);
        gemm128(xln, DMODEL, 0, wqkvT + 2 * DMODEL * DMODEL, DMODEL, 0, nullptr, vbf, DMODEL, 0,
                nullptr, nullptr, 0, N_TOK, DMODEL, DMODEL, 1, 1, 1.0f, 0, 0);

        // landmarks + v transpose (v^T is the B-operand of a3@v)
        k_landmarks<<<(NHEAD * MLM * DHEAD) / 256, 256, 0, stream>>>(qbf, kbf, qlbf, klbf);
        k_transpose<<<dim3(N_TOK / 32, DMODEL / 32), 256, 0, stream>>>(vbf, vT);

        // s1 = q @ k_l^T  (B^T = k_l, per head, K=64) -> softmax -> a1 (bf16)
        gemm128(qbf, DMODEL, DHEAD, klbf, DHEAD, (long long)MLM * DHEAD,
                scores, nullptr, MLM, (long long)N_TOK * MLM,
                nullptr, nullptr, 0, N_TOK, MLM, DHEAD, NHEAD, 1, 1.0f, 0, 0);
        k_softmax<<<NHEAD * N_TOK, 256, 0, stream>>>(scores, a1bf, nullptr, MLM);

        // s2 = q_l @ k_l^T -> softmax -> a2 (f32)
        gemm128(qlbf, DHEAD, (long long)MLM * DHEAD, klbf, DHEAD, (long long)MLM * DHEAD,
                s2, nullptr, MLM, (long long)MLM * MLM,
                nullptr, nullptr, 0, MLM, MLM, DHEAD, NHEAD, 1, 1.0f, 0, 0);
        k_softmax<<<NHEAD * MLM, 256, 0, stream>>>(s2, nullptr, a2, MLM);

        // s3 = q_l @ k^T (B^T = k itself, per head) -> softmax -> a3 (bf16)
        gemm128(qlbf, DHEAD, (long long)MLM * DHEAD, kbf, DMODEL, DHEAD,
                scores, nullptr, N_TOK, (long long)MLM * N_TOK,
                nullptr, nullptr, 0, MLM, N_TOK, DHEAD, NHEAD, 1, 1.0f, 0, 0);
        k_softmax<<<NHEAD * MLM, 256, 0, stream>>>(scores, a3bf, nullptr, N_TOK);

        // Newton-Schulz pinv of a2 (fp32)
        k_pinv_init<<<NHEAD, 256, 0, stream>>>(a2, zA);
        float* zc = zA; float* zn = zB;
        for (int it = 0; it < 6; ++it) {
            dim3 g(4, 4, NHEAD);
            k_mm256<<<g, 256, 0, stream>>>(a2, zc, xz, 0.0f, 1.0f);
            k_diag_axpy<<<(NHEAD * MLM * MLM) / 256, 256, 0, stream>>>(xz, y1, 15.0f, -1.0f);
            k_mm256<<<g, 256, 0, stream>>>(xz, y1, y2, 28.0f, -1.0f);
            k_mm256<<<g, 256, 0, stream>>>(xz, y2, y3, 13.0f, -1.0f);
            k_mm256<<<g, 256, 0, stream>>>(zc, y3, zn, 0.0f, 0.25f);
            float* t = zc; zc = zn; zn = t;
        }

        // a3v = a3 @ v  (B^T = v^T, per head, K=16384, split-K atomic accumulate)
        k_zero_f32<<<(NHEAD * MLM * DHEAD) / 256, 256, 0, stream>>>(a3v, NHEAD * MLM * DHEAD);
        gemm64(a3bf, N_TOK, (long long)MLM * N_TOK, vT, N_TOK, (long long)DHEAD * N_TOK,
               a3v, nullptr, DHEAD, (long long)MLM * DHEAD,
               nullptr, nullptr, 0, MLM, DHEAD, N_TOK, NHEAD, 16, 1.0f, 0, 0);

        // w2 = pinv(a2) @ a3v  (B^T = a3v^T; output written transposed -> w2T)
        cvt(zc, zbf, (long long)NHEAD * MLM * MLM);
        cvtT(a3v, a3vT, MLM, DHEAD, NHEAD);          // [h][64][256]
        gemm64(zbf, MLM, (long long)MLM * MLM, a3vT, MLM, (long long)DHEAD * MLM,
               nullptr, w2T, MLM, (long long)DHEAD * MLM,
               nullptr, nullptr, 0, MLM, DHEAD, MLM, NHEAD, 1, 1.0f, 0, /*cbT=*/1);

        // attn = a1 @ w2 (B^T = w2T, per head into its 64 columns of attn[N,512])
        gemm64(a1bf, MLM, (long long)N_TOK * MLM, w2T, MLM, (long long)DHEAD * MLM,
               attn, nullptr, DMODEL, (long long)DHEAD,
               nullptr, nullptr, 0, N_TOK, DHEAD, MLM, NHEAD, 1, 1.0f, 0, 0);

        // attn += depthwise_conv(v)
        k_resconv<<<(N_TOK * DMODEL) / 256, 256, 0, stream>>>(vbf, cw, attn);

        // x = x + attn @ w_out + b_out
        cvt(attn, attnbf, (long long)N_TOK * DMODEL);
        gemm128(attnbf, DMODEL, 0, woutT, DMODEL, 0, x, nullptr, DMODEL, 0,
                bout, x, DMODEL, N_TOK, DMODEL, DMODEL, 1, 1, 1.0f, 0, 0);

        // FFN: x = x + gelu(LN2(x) @ w_ff1 + b_ff1) @ w_ff2 + b_ff2
        k_layernorm<<<N_TOK, 256, 0, stream>>>(x, ln2w, ln2b, xln);
        gemm128(xln, DMODEL, 0, wff1T, DMODEL, 0, nullptr, ffbf, FFDIM, 0,
                bff1, nullptr, 0, N_TOK, FFDIM, DMODEL, 1, 1, 1.0f, 1, 0);
        gemm128(ffbf, FFDIM, 0, wff2T, FFDIM, 0, x, nullptr, DMODEL, 0,
                bff2, x, DMODEL, N_TOK, DMODEL, FFDIM, 1, 1, 1.0f, 0, 0);
    }

    // output: x[:, :N_P]
    hipMemcpyAsync(d_out, x, (size_t)N_P * DMODEL * 4, hipMemcpyDeviceToDevice, stream);
}
